// Llama_46986942218422
// MI455X (gfx1250) — compile-verified, matrix operand order checked by reading, not run on hardware
//
#include <hip/hip_runtime.h>

typedef __bf16 v16bf __attribute__((ext_vector_type(16)));
typedef float  v8f  __attribute__((ext_vector_type(8)));
typedef unsigned int v4u __attribute__((ext_vector_type(4)));
typedef int v4i __attribute__((ext_vector_type(4)));
typedef int v8i __attribute__((ext_vector_type(8)));

#define TOKENS 2048
#define DMODEL 768
#define NHEAD  12
#define HSZ    64
#define KVW    192   // NKV*HS
#define FFI    2048
#define NLAYER 12
#define SEQ    1024
#define VOCAB  50257

#if __has_builtin(__builtin_amdgcn_tensor_load_to_lds)
#define GEMM_TDM 1
#else
#define GEMM_TDM 0
#endif

union FragA { unsigned int u[8]; v16bf v; };
union FragC { float f[8]; v8f v; };

__device__ __forceinline__ unsigned short f2bf(float x){
  unsigned int u = __float_as_uint(x);
  unsigned int r = u + 0x7FFFu + ((u >> 16) & 1u);   // round-to-nearest-even
  return (unsigned short)(r >> 16);
}

// K-index of the low element of VGPR-pair v for the 16-bit WMMA A/B layout
// (ISA 7.12.2: lanes 0-15 hold K 0..7 / 16..23, lanes 16-31 hold K 8..15 / 24..31)
__device__ __forceinline__ int frag_k(int v, int hi){
  return ((v < 4) ? (2*v) : (16 + 2*(v-4))) + hi*8;
}

// butterfly reductions across 16-lane halves via ds_swizzle_b32 (xor mode)
__device__ __forceinline__ float red_max16(float x){
  x = fmaxf(x, __uint_as_float((unsigned)__builtin_amdgcn_ds_swizzle((int)__float_as_uint(x), 0x041F)));
  x = fmaxf(x, __uint_as_float((unsigned)__builtin_amdgcn_ds_swizzle((int)__float_as_uint(x), 0x081F)));
  x = fmaxf(x, __uint_as_float((unsigned)__builtin_amdgcn_ds_swizzle((int)__float_as_uint(x), 0x101F)));
  x = fmaxf(x, __uint_as_float((unsigned)__builtin_amdgcn_ds_swizzle((int)__float_as_uint(x), 0x201F)));
  return x;
}
__device__ __forceinline__ float red_sum16(float x){
  x += __uint_as_float((unsigned)__builtin_amdgcn_ds_swizzle((int)__float_as_uint(x), 0x041F));
  x += __uint_as_float((unsigned)__builtin_amdgcn_ds_swizzle((int)__float_as_uint(x), 0x081F));
  x += __uint_as_float((unsigned)__builtin_amdgcn_ds_swizzle((int)__float_as_uint(x), 0x101F));
  x += __uint_as_float((unsigned)__builtin_amdgcn_ds_swizzle((int)__float_as_uint(x), 0x201F));
  return x;
}

// ---------------------------------------------------------------- embedding
__global__ void embed_kernel(const int* __restrict__ idx,
                             const float* __restrict__ embed,
                             float* __restrict__ x){
  int tok = blockIdx.x;
  int id  = idx[tok];
  for (int d = threadIdx.x; d < DMODEL; d += blockDim.x)
    x[(size_t)tok*DMODEL + d] = embed[(size_t)id*DMODEL + d];
}

// ---------------------------------------------------------------- rmsnorm -> bf16
__global__ void rmsnorm_kernel(const float* __restrict__ x,
                               const float* __restrict__ w,
                               unsigned short* __restrict__ hbf){
  int tok = blockIdx.x;
  const float* row = x + (size_t)tok*DMODEL;
  int tid = threadIdx.x;
  float v0 = row[tid], v1 = row[tid+256], v2 = row[tid+512];
  float s = v0*v0 + v1*v1 + v2*v2;
  for (int off = 16; off; off >>= 1) s += __shfl_xor(s, off, 32);
  __shared__ float red[8];
  __shared__ float rrms;
  int wid = tid >> 5, lane = tid & 31;
  if (lane == 0) red[wid] = s;
  __syncthreads();
  if (tid == 0){
    float t = 0.f;
    for (int i = 0; i < 8; i++) t += red[i];
    rrms = rsqrtf(t / (float)DMODEL + 1e-6f);
  }
  __syncthreads();
  float r = rrms;
  unsigned short* ho = hbf + (size_t)tok*DMODEL;
  ho[tid]     = f2bf(v0 * r * w[tid]);
  ho[tid+256] = f2bf(v1 * r * w[tid+256]);
  ho[tid+512] = f2bf(v2 * r * w[tid+512]);
}

// ---------------------------------------------------------------- WMMA GEMM
// C[M,N] (+)= A_bf16[M,K] @ W_f32[K,N]   (BT: W stored [N,K], guarded for N%64)
// 256 threads = 8 waves. Block tile 128(M) x 64(N), K staged 64 at a time.
// Wave grid 4(M) x 2(N): each wave owns 32x32 = 2x2 WMMA tiles -> 8 WMMA/stage,
// A frags reused across 2 N-tiles, B frags across 2 M-tiles.
// A tile staged to LDS by the Tensor Data Mover (wave 0 issues, TENSORcnt sync).
template<int RESID, int BT>
__global__ void __launch_bounds__(256, 1)
gemm_kernel(const unsigned short* __restrict__ A,
            const float* __restrict__ W,
            float* __restrict__ C,
            int N, int K){
  __shared__ __align__(16) unsigned short As[128*64];
  __shared__ __align__(16) unsigned short Bs[64*64];
  int tid  = threadIdx.x;
  int lane = tid & 31, wid = tid >> 5;
  int hi = lane >> 4, lo = lane & 15;
  int wm = wid >> 1, wn = wid & 1;      // 4 x 2 wave grid
  int m0 = blockIdx.y * 128;
  int n0 = blockIdx.x * 64;

  FragC acc[2][2];
  #pragma unroll
  for (int i = 0; i < 2; i++)
    #pragma unroll
    for (int j = 0; j < 2; j++)
      #pragma unroll
      for (int v = 0; v < 8; v++) acc[i][j].f[v] = 0.f;

  for (int k0 = 0; k0 < K; k0 += 64){
#if GEMM_TDM
    if (wid == 0){
      // Tensor DMA: 128x64 bf16 tile of A (row stride K elements) -> As
      unsigned long long ga = (unsigned long long)(const void*)(A + (size_t)m0*K + k0);
      v4u g0;
      g0[0] = 1u;                                            // count=1, user mode
      g0[1] = (unsigned)(size_t)(void*)As;                   // LDS byte address
      g0[2] = (unsigned)(ga & 0xFFFFFFFFull);                // global_addr[31:0]
      g0[3] = (unsigned)((ga >> 32) & 0x01FFFFFFull) | (2u << 30); // addr[56:32] | type=2
      v8i g1;
      g1[0] = (1 << 16);          // workgroup_mask=0, data_size=1 (2 bytes)
      g1[1] = (64 << 16);         // tensor_dim0[15:0]=64 (bits 79:48 low half)
      g1[2] = (128 << 16);        // tensor_dim0 hi=0 | tensor_dim1[15:0]=128
      g1[3] = (64 << 16);         // tensor_dim1 hi=0 | tile_dim0=64
      g1[4] = 128;                // tile_dim1=128, tile_dim2=0
      g1[5] = K;                  // tensor_dim0_stride[31:0] = K elements
      g1[6] = 0;                  // stride hi, tensor_dim1_stride lo
      g1[7] = 0;
      v4i gz4 = {0, 0, 0, 0};
      v8i gz8 = {0, 0, 0, 0, 0, 0, 0, 0};
      __builtin_amdgcn_tensor_load_to_lds(g0, g1, gz4, gz4, gz8, 0);
    }
#else
    { // manual A staging: 128x64 halves, 32 per thread (4 x b128)
      int f = tid * 32; int r = f >> 6, c = f & 63;
      const uint4* src = (const uint4*)(A + (size_t)(m0+r)*K + k0 + c);
      uint4* dst = (uint4*)&As[f];
      #pragma unroll
      for (int j = 0; j < 4; j++) dst[j] = src[j];
    }
#endif
    { // stage B tile (f32 -> bf16, 64x64): 16 floats per thread
      int f = tid * 16; int r = f >> 6, c = f & 63;
      if (!BT){
        const float4* p0 = (const float4*)(W + (size_t)(k0+r)*N + n0 + c);
        #pragma unroll
        for (int j = 0; j < 4; j++){
          float4 a = p0[j];
          Bs[r*64+c+4*j+0] = f2bf(a.x);
          Bs[r*64+c+4*j+1] = f2bf(a.y);
          Bs[r*64+c+4*j+2] = f2bf(a.z);
          Bs[r*64+c+4*j+3] = f2bf(a.w);
        }
      } else {
        #pragma unroll
        for (int i = 0; i < 16; i++){
          int gcol = n0 + c + i;
          float vv = (gcol < N) ? W[(size_t)gcol*K + k0 + r] : 0.f;
          Bs[r*64+c+i] = f2bf(vv);
        }
      }
    }
#if GEMM_TDM
    if (wid == 0) __builtin_amdgcn_s_wait_tensorcnt(0);
#endif
    __syncthreads();
    #pragma unroll
    for (int kk = 0; kk < 64; kk += 32){
      FragA a0, a1, b0, b1;
      #pragma unroll
      for (int v = 0; v < 8; v++){
        int k = frag_k(v, hi) + kk;
        a0.u[v] = *(const unsigned int*)&As[(wm*32      + lo)*64 + k];
        a1.u[v] = *(const unsigned int*)&As[(wm*32 + 16 + lo)*64 + k];
        unsigned int e0 = Bs[ k   *64 + wn*32 + lo];
        unsigned int e1 = Bs[(k+1)*64 + wn*32 + lo];
        b0.u[v] = e0 | (e1 << 16);
        e0 = Bs[ k   *64 + wn*32 + 16 + lo];
        e1 = Bs[(k+1)*64 + wn*32 + 16 + lo];
        b1.u[v] = e0 | (e1 << 16);
      }
      acc[0][0].v = __builtin_amdgcn_wmma_f32_16x16x32_bf16(false, a0.v, false, b0.v, (short)0, acc[0][0].v, false, false);
      acc[0][1].v = __builtin_amdgcn_wmma_f32_16x16x32_bf16(false, a0.v, false, b1.v, (short)0, acc[0][1].v, false, false);
      acc[1][0].v = __builtin_amdgcn_wmma_f32_16x16x32_bf16(false, a1.v, false, b0.v, (short)0, acc[1][0].v, false, false);
      acc[1][1].v = __builtin_amdgcn_wmma_f32_16x16x32_bf16(false, a1.v, false, b1.v, (short)0, acc[1][1].v, false, false);
    }
    __syncthreads();
  }
  #pragma unroll
  for (int i = 0; i < 2; i++)
    #pragma unroll
    for (int j = 0; j < 2; j++)
      #pragma unroll
      for (int v = 0; v < 8; v++){
        int row = m0 + wm*32 + i*16 + v + hi*8;
        int col = n0 + wn*32 + j*16 + lo;
        if (!BT || col < N){
          size_t o = (size_t)row * N + col;
          C[o] = (RESID ? C[o] : 0.f) + acc[i][j].f[v];
        }
      }
}

// ---------------------------------------------------------------- RoPE + bf16 convert
// Folds the softmax 1/sqrt(HS) scale into Q (rotation is linear).
__global__ void rope_cvt_kernel(const float* __restrict__ qf,
                                const float* __restrict__ kf,
                                const float* __restrict__ vf,
                                unsigned short* __restrict__ qb,
                                unsigned short* __restrict__ kb,
                                unsigned short* __restrict__ vb){
  int tok = blockIdx.x;
  int t = tok & (SEQ - 1);
  for (int s = threadIdx.x; s < DMODEL + 2*KVW; s += blockDim.x){
    if (s < DMODEL){
      int hd = s & 63, i = hd & 31;
      float ang = (float)t * __expf(-(float)i * 0.28782313662f); // 10000^(-i/32)
      float sn = __sinf(ang), cs = __cosf(ang);
      const float* src = qf + (size_t)tok*DMODEL;
      float xv = src[s];
      float partner = (hd < 32) ? -src[s+32] : src[s-32];
      qb[(size_t)tok*DMODEL + s] = f2bf((xv*cs + partner*sn) * 0.125f);
    } else if (s < DMODEL + KVW){
      int d = s - DMODEL;
      int hd = d & 63, i = hd & 31;
      float ang = (float)t * __expf(-(float)i * 0.28782313662f);
      float sn = __sinf(ang), cs = __cosf(ang);
      const float* src = kf + (size_t)tok*KVW;
      float xv = src[d];
      float partner = (hd < 32) ? -src[d+32] : src[d-32];
      kb[(size_t)tok*KVW + d] = f2bf(xv*cs + partner*sn);
    } else {
      int d = s - DMODEL - KVW;
      vb[(size_t)tok*KVW + d] = f2bf(vf[(size_t)tok*KVW + d]);
    }
  }
}

// ---------------------------------------------------------------- flash attention
// 1 wave per (b, head, 16-query tile). 32-key blocks, causal, online softmax.
// launch_bounds(32,1): single wave per block, allow max VGPRs -> no spills.
__global__ void __launch_bounds__(32, 1)
attn_kernel(const unsigned short* __restrict__ qb,
            const unsigned short* __restrict__ kb,
            const unsigned short* __restrict__ vb,
            unsigned short* __restrict__ yb){
  __shared__ __align__(16) float P[16][32];
  int q0 = blockIdx.x * 16;
  int h  = blockIdx.y;
  int b  = blockIdx.z;
  int kvh = h >> 2;                    // GQA: REP = 4
  int lane = threadIdx.x;
  int hi = lane >> 4, lo = lane & 15;

  FragA qa0, qa1;
  #pragma unroll
  for (int v = 0; v < 8; v++){
    int k = frag_k(v, hi);
    const unsigned short* p = qb + (size_t)(b*SEQ + q0 + lo)*DMODEL + h*HSZ;
    qa0.u[v] = *(const unsigned int*)(p + k);
    qa1.u[v] = *(const unsigned int*)(p + 32 + k);
  }

  float mrow[8], lrow[8];
  FragC o[4];
  #pragma unroll
  for (int v = 0; v < 8; v++){ mrow[v] = -1e30f; lrow[v] = 0.f; }
  #pragma unroll
  for (int t = 0; t < 4; t++)
    #pragma unroll
    for (int v = 0; v < 8; v++) o[t].f[v] = 0.f;

  int kbmax = (q0 + 15) >> 5;
  for (int kbi = 0; kbi <= kbmax; kbi++){
    int j0 = kbi * 32;
    FragA b0lo, b0hi, b1lo, b1hi;      // K^T fragments (dims x keys)
    #pragma unroll
    for (int v = 0; v < 8; v++){
      int k = frag_k(v, hi);
      const unsigned short* p0 = kb + (size_t)(b*SEQ + j0 + lo)*KVW + kvh*HSZ;
      const unsigned short* p1 = kb + (size_t)(b*SEQ + j0 + 16 + lo)*KVW + kvh*HSZ;
      b0lo.u[v] = *(const unsigned int*)(p0 + k);
      b0hi.u[v] = *(const unsigned int*)(p0 + 32 + k);
      b1lo.u[v] = *(const unsigned int*)(p1 + k);
      b1hi.u[v] = *(const unsigned int*)(p1 + 32 + k);
    }
    FragC s0, s1;
    #pragma unroll
    for (int v = 0; v < 8; v++){ s0.f[v] = 0.f; s1.f[v] = 0.f; }
    s0.v = __builtin_amdgcn_wmma_f32_16x16x32_bf16(false, qa0.v, false, b0lo.v, (short)0, s0.v, false, false);
    s0.v = __builtin_amdgcn_wmma_f32_16x16x32_bf16(false, qa1.v, false, b0hi.v, (short)0, s0.v, false, false);
    s1.v = __builtin_amdgcn_wmma_f32_16x16x32_bf16(false, qa0.v, false, b1lo.v, (short)0, s1.v, false, false);
    s1.v = __builtin_amdgcn_wmma_f32_16x16x32_bf16(false, qa1.v, false, b1hi.v, (short)0, s1.v, false, false);

    float alpha[8];
    #pragma unroll
    for (int v = 0; v < 8; v++){
      int r  = v + hi*8;               // C-layout: row = vgpr + 8*(lane>=16)
      int qi = q0 + r;
      float sv0 = s0.f[v];             // scale pre-folded into Q
      float sv1 = s1.f[v];
      if (j0 + lo      > qi) sv0 = -1e30f;
      if (j0 + 16 + lo > qi) sv1 = -1e30f;
      float mx  = red_max16(fmaxf(sv0, sv1));
      float mn  = fmaxf(mrow[v], mx);
      float al  = __expf(mrow[v] - mn);
      float p0v = __expf(sv0 - mn);
      float p1v = __expf(sv1 - mn);
      float rs  = red_sum16(p0v + p1v);
      lrow[v] = lrow[v]*al + rs;
      mrow[v] = mn;
      alpha[v] = al;
      P[r][lo]      = p0v;
      P[r][16 + lo] = p1v;
    }
    __syncthreads();
    #pragma unroll
    for (int t = 0; t < 4; t++)
      #pragma unroll
      for (int v = 0; v < 8; v++) o[t].f[v] *= alpha[v];

    FragA pa;                           // P re-laid out as A fragment (M=rows, K=keys)
    #pragma unroll
    for (int v = 0; v < 8; v++){
      int k = frag_k(v, hi);
      unsigned int e0 = f2bf(P[lo][k]);
      unsigned int e1 = f2bf(P[lo][k+1]);
      pa.u[v] = e0 | (e1 << 16);
    }
    #pragma unroll
    for (int t = 0; t < 4; t++){
      FragA vbf;                        // V fragment (keys x 16 head-dims)
      #pragma unroll
      for (int v = 0; v < 8; v++){
        int k = frag_k(v, hi);
        unsigned int e0 = vb[(size_t)(b*SEQ + j0 + k    )*KVW + kvh*HSZ + t*16 + lo];
        unsigned int e1 = vb[(size_t)(b*SEQ + j0 + k + 1)*KVW + kvh*HSZ + t*16 + lo];
        vbf.u[v] = e0 | (e1 << 16);
      }
      o[t].v = __builtin_amdgcn_wmma_f32_16x16x32_bf16(false, pa.v, false, vbf.v,
                                                       (short)0, o[t].v, false, false);
    }
    __syncthreads();
  }
  #pragma unroll
  for (int t = 0; t < 4; t++)
    #pragma unroll
    for (int v = 0; v < 8; v++){
      int r = v + hi*8;
      float val = o[t].f[v] / lrow[v];
      yb[(size_t)(b*SEQ + q0 + r)*DMODEL + h*HSZ + t*16 + lo] = f2bf(val);
    }
}

// ---------------------------------------------------------------- SwiGLU elementwise
__global__ void silu_mul_kernel(const float* __restrict__ g,
                                const float* __restrict__ u,
                                unsigned short* __restrict__ gu, int n){
  for (int i = blockIdx.x*blockDim.x + threadIdx.x; i < n; i += gridDim.x*blockDim.x){
    float gv = g[i];
    float s  = gv / (1.f + __expf(-gv));
    gu[i] = f2bf(s * u[i]);
  }
}

// ---------------------------------------------------------------- host
extern "C" void kernel_launch(void* const* d_in, const int* in_sizes, int n_in,
                              void* d_out, int out_size, void* d_ws, size_t ws_size,
                              hipStream_t stream){
  (void)in_sizes; (void)n_in; (void)out_size; (void)ws_size;
  const int*   idx   = (const int*)  d_in[0];
  const float* embed = (const float*)d_in[1];
  const float* ln1   = (const float*)d_in[2];
  const float* qw    = (const float*)d_in[3];
  const float* kw    = (const float*)d_in[4];
  const float* vw    = (const float*)d_in[5];
  const float* ow    = (const float*)d_in[6];
  const float* ln2   = (const float*)d_in[7];
  const float* gw    = (const float*)d_in[8];
  const float* uw    = (const float*)d_in[9];
  const float* dw    = (const float*)d_in[10];
  const float* nw    = (const float*)d_in[11];
  float* out = (float*)d_out;

  char* ws = (char*)d_ws;
  size_t off = 0;
  auto alloc = [&](size_t bytes)->void*{
    void* p = ws + off; off += (bytes + 255) & ~(size_t)255; return p;
  };
  float*          x   = (float*)         alloc((size_t)TOKENS*DMODEL*4);
  unsigned short* hbf = (unsigned short*)alloc((size_t)TOKENS*DMODEL*2);
  float*          qf  = (float*)         alloc((size_t)TOKENS*DMODEL*4);
  float*          kf  = (float*)         alloc((size_t)TOKENS*KVW*4);
  float*          vf  = (float*)         alloc((size_t)TOKENS*KVW*4);
  unsigned short* qbp = (unsigned short*)alloc((size_t)TOKENS*DMODEL*2);
  unsigned short* kbp = (unsigned short*)alloc((size_t)TOKENS*KVW*2);
  unsigned short* vbp = (unsigned short*)alloc((size_t)TOKENS*KVW*2);
  unsigned short* yb  = (unsigned short*)alloc((size_t)TOKENS*DMODEL*2);
  float*          gf  = (float*)         alloc((size_t)TOKENS*FFI*4);
  float*          uf  = (float*)         alloc((size_t)TOKENS*FFI*4);
  unsigned short* gub = (unsigned short*)alloc((size_t)TOKENS*FFI*2);

  embed_kernel<<<TOKENS, 256, 0, stream>>>(idx, embed, x);
  for (int l = 0; l < NLAYER; l++){
    rmsnorm_kernel<<<TOKENS, 256, 0, stream>>>(x, ln1 + (size_t)l*DMODEL, hbf);
    gemm_kernel<0,0><<<dim3(DMODEL/64, TOKENS/128), 256, 0, stream>>>(hbf, qw + (size_t)l*DMODEL*DMODEL, qf, DMODEL, DMODEL);
    gemm_kernel<0,0><<<dim3(KVW/64,    TOKENS/128), 256, 0, stream>>>(hbf, kw + (size_t)l*DMODEL*KVW,   kf, KVW,    DMODEL);
    gemm_kernel<0,0><<<dim3(KVW/64,    TOKENS/128), 256, 0, stream>>>(hbf, vw + (size_t)l*DMODEL*KVW,   vf, KVW,    DMODEL);
    rope_cvt_kernel<<<TOKENS, 256, 0, stream>>>(qf, kf, vf, qbp, kbp, vbp);
    attn_kernel<<<dim3(SEQ/16, NHEAD, 2), 32, 0, stream>>>(qbp, kbp, vbp, yb);
    gemm_kernel<1,0><<<dim3(DMODEL/64, TOKENS/128), 256, 0, stream>>>(yb, ow + (size_t)l*DMODEL*DMODEL, x, DMODEL, DMODEL);
    rmsnorm_kernel<<<TOKENS, 256, 0, stream>>>(x, ln2 + (size_t)l*DMODEL, hbf);
    gemm_kernel<0,0><<<dim3(FFI/64, TOKENS/128), 256, 0, stream>>>(hbf, gw + (size_t)l*DMODEL*FFI, gf, FFI, DMODEL);
    gemm_kernel<0,0><<<dim3(FFI/64, TOKENS/128), 256, 0, stream>>>(hbf, uw + (size_t)l*DMODEL*FFI, uf, FFI, DMODEL);
    silu_mul_kernel<<<512, 256, 0, stream>>>(gf, uf, gub, TOKENS*FFI);
    gemm_kernel<1,0><<<dim3(DMODEL/64, TOKENS/128), 256, 0, stream>>>(gub, dw + (size_t)l*FFI*DMODEL, x, DMODEL, FFI);
  }
  rmsnorm_kernel<<<TOKENS, 256, 0, stream>>>(x, nw, hbf);
  gemm_kernel<0,1><<<dim3((VOCAB+63)/64, TOKENS/128), 256, 0, stream>>>(hbf, embed, out, VOCAB, DMODEL);
}
